// TextureBlock_31799937859957
// MI455X (gfx1250) — 8-bit matrix operand order repaired and checked by reading, compiled, not run on hardware
//
#include <hip/hip_runtime.h>
#include <hip/hip_bf16.h>

typedef int v8i __attribute__((ext_vector_type(8)));

#define NB   8
#define NC   3
#define NH   512
#define NW   512
#define KS   5
#define PAD  2

__device__ __forceinline__ int reflect101(int i, int n) {
    i = (i < 0) ? -i : i;
    i = (i >= n) ? (2 * n - 2 - i) : i;
    return i;
}

__device__ __forceinline__ int gray_level(const float* __restrict__ x,
                                          int b, int y, int xx) {
    const int plane = NH * NW;
    const int base  = (b * NC) * plane + y * NW + xx;
    float r = rintf(x[base] * 255.0f);
    float g = rintf(x[base + plane] * 255.0f);
    float bl = rintf(x[base + 2 * plane] * 255.0f);
    r  = fminf(fmaxf(r, 0.0f), 255.0f);
    g  = fminf(fmaxf(g, 0.0f), 255.0f);
    bl = fminf(fmaxf(bl, 0.0f), 255.0f);
    float gy = rintf(r * 0.299f + g * 0.587f + bl * 0.114f);
    gy = fminf(fmaxf(gy, 0.0f), 255.0f);
    int gl = ((int)gy) >> 5;
    return gl;
}

__global__ __launch_bounds__(256)
void glcm_code_kernel(const float* __restrict__ x, unsigned char* __restrict__ code) {
    int idx = blockIdx.x * 256 + threadIdx.x;
    const int total = NB * NH * NW;
    if (idx >= total) return;
    int b   = idx / (NH * NW);
    int rem = idx - b * (NH * NW);
    int y   = rem / NW;
    int xx  = rem - y * NW;
    int g1 = gray_level(x, b, y, xx);
    int x2 = (xx + 1 < NW) ? (xx + 1) : (NW - 1);
    int g2 = gray_level(x, b, y, x2);
    code[idx] = (unsigned char)(g1 * 8 + g2);
}

__global__ __launch_bounds__(256)
void glcm_asm_kernel(const unsigned char* __restrict__ code,
                     float* __restrict__ out) {
    const int lane    = threadIdx.x & 31;
    const int waveIdx = (blockIdx.x * (256 / 32)) + (threadIdx.x >> 5);

    const int wavesPerRow = NW / 16;
    const int x0 = (waveIdx % wavesPerRow) * 16;
    const int y  = (waveIdx / wavesPerRow) % NH;
    const int b  =  waveIdx / (wavesPerRow * NH);

    const unsigned hi = (unsigned)(lane >> 4);
    const int m  = lane & 15;
    const int xc = x0 + m;

    int cols[KS];
#pragma unroll
    for (int d = 0; d < KS; ++d) cols[d] = reflect101(xc + d - PAD, NW);

    unsigned h0 = 0, h1 = 0, h2 = 0, h3 = 0, h4 = 0, h5 = 0, h6 = 0, h7 = 0;

#pragma unroll
    for (int dy = 0; dy < KS; ++dy) {
        const int ry   = reflect101(y + dy - PAD, NH);
        const int base = (b * NH + ry) * NW;
#pragma unroll
        for (int dx = 0; dx < KS; ++dx) {
            const unsigned c   = (unsigned)code[base + cols[dx]];
            const unsigned sel = c >> 2;
            const unsigned own = (((c >> 3) & 1u) == hi) ? 1u : 0u;
            const unsigned inc = own ? (1u << ((c & 3u) << 3)) : 0u;
            const unsigned vg  = ((sel >> 2) << 1) | (sel & 1u);
            h0 += (vg == 0u) ? inc : 0u;
            h1 += (vg == 1u) ? inc : 0u;
            h2 += (vg == 2u) ? inc : 0u;
            h3 += (vg == 3u) ? inc : 0u;
            h4 += (vg == 4u) ? inc : 0u;
            h5 += (vg == 5u) ? inc : 0u;
            h6 += (vg == 6u) ? inc : 0u;
            h7 += (vg == 7u) ? inc : 0u;
        }
    }

    const int SWAPX16 = 0x401F;
    unsigned o0 = (unsigned)__builtin_amdgcn_ds_swizzle((int)h0, SWAPX16);
    unsigned o1 = (unsigned)__builtin_amdgcn_ds_swizzle((int)h1, SWAPX16);
    unsigned o2 = (unsigned)__builtin_amdgcn_ds_swizzle((int)h2, SWAPX16);
    unsigned o3 = (unsigned)__builtin_amdgcn_ds_swizzle((int)h3, SWAPX16);
    unsigned o4 = (unsigned)__builtin_amdgcn_ds_swizzle((int)h4, SWAPX16);
    unsigned o5 = (unsigned)__builtin_amdgcn_ds_swizzle((int)h5, SWAPX16);
    unsigned o6 = (unsigned)__builtin_amdgcn_ds_swizzle((int)h6, SWAPX16);
    unsigned o7 = (unsigned)__builtin_amdgcn_ds_swizzle((int)h7, SWAPX16);

    v8i a;
    a[0] = (int)h0; a[1] = (int)h1; a[2] = (int)h2; a[3] = (int)h3;
    a[4] = (int)h4; a[5] = (int)h5; a[6] = (int)h6; a[7] = (int)h7;

    v8i bb;
    bb[0] = (int)h0;
    bb[1] = (int)h1;
    bb[2] = (int)h2;
    bb[3] = (int)h3;
    bb[4] = (int)h4;
    bb[5] = (int)h5;
    bb[6] = (int)h6;
    bb[7] = (int)h7;

    v8i cz = {};
    v8i d = __builtin_amdgcn_wmma_i32_16x16x64_iu8(
         false, a,  false, bb, cz,
         false,  false);

    const bool active = (lane < 8) || (lane >= 24);
    const int  idx    = lane & 7;
    int val = d[0];
    if (idx == 1) val = d[1];
    if (idx == 2) val = d[2];
    if (idx == 3) val = d[3];
    if (idx == 4) val = d[4];
    if (idx == 5) val = d[5];
    if (idx == 6) val = d[6];
    if (idx == 7) val = d[7];

    if (active) {
        const int pm = (lane < 8) ? lane : (lane - 16);
        out[(b * NH + y) * NW + x0 + pm] = (float)val;
    }
}

extern "C" void kernel_launch(void* const* d_in, const int* in_sizes, int n_in,
                              void* d_out, int out_size, void* d_ws, size_t ws_size,
                              hipStream_t stream) {
    const float*   x    = (const float*)d_in[0];
    float*         out  = (float*)d_out;
    unsigned char* code = (unsigned char*)d_ws;

    const int totalPix = NB * NH * NW;
    glcm_code_kernel<<<(totalPix + 255) / 256, 256, 0, stream>>>(x, code);

    const int totalWaves = totalPix / 16;
    glcm_asm_kernel<<<totalWaves / 8, 256, 0, stream>>>(code, out);
}
